// DQN_37718402793615
// MI455X (gfx1250) — matrix operand order repaired and checked by reading, not compiled, not run on hardware
//
#include <hip/hip_runtime.h>
#include <hip/hip_bf16.h>
#include <math.h>

typedef __attribute__((ext_vector_type(16))) _Float16 v16h;
typedef __attribute__((ext_vector_type(8)))  float    v8f;

#define NEG_SLOPE 0.2f

__device__ __forceinline__ float sigmoidf_(float x) {
    return 1.0f / (1.0f + __expf(-x));
}

__device__ __forceinline__ void atomicMaxFloat(float* addr, float val) {
    unsigned int* ua = (unsigned int*)addr;
    unsigned int old = *ua;
    while (__uint_as_float(old) < val) {
        unsigned int assumed = old;
        old = atomicCAS(ua, assumed, __float_as_uint(val));
        if (old == assumed) break;
    }
}

// ---------------------------------------------------------------------------
// Init: zero the accumulation region, set segment-max to -FLT_MAX.
// ---------------------------------------------------------------------------
__global__ void k_init(float* __restrict__ zero_base, long nzero,
                       float* __restrict__ smax, int n) {
    long i = (long)blockIdx.x * blockDim.x + threadIdx.x;
    if (i < nzero) zero_base[i] = 0.0f;
    if (i < n)     smax[i] = -3.402823466e38f;
}

// deg[dst] += 1 ; loop_sum[dst] += edge_attr  (self-loop fill_value='mean')
__global__ void k_edge_pre(const int* __restrict__ ei, const float* __restrict__ ea,
                           float* __restrict__ deg, float* __restrict__ loop_sum, int E) {
    int e = blockIdx.x * blockDim.x + threadIdx.x;
    if (e >= E) return;
    int d = ei[E + e];
    atomicAdd(&deg[d], 1.0f);
    atomicAdd(&loop_sum[d], ea[e]);
}

__global__ void k_pool_count(const int* __restrict__ pb, float* __restrict__ cnt, int N) {
    int i = blockIdx.x * blockDim.x + threadIdx.x;
    if (i >= N) return;
    atomicAdd(&cnt[pb[i]], 1.0f);
}

__global__ void k_loop_fin(float* __restrict__ loop_sum, const float* __restrict__ deg, int N) {
    int i = blockIdx.x * blockDim.x + threadIdx.x;
    if (i >= N) return;
    loop_sum[i] = loop_sum[i] / fmaxf(deg[i], 1.0f);
}

// ---------------------------------------------------------------------------
// Pre-pack B operands (W_l, W_r, W_fc1) into WMMA 32x16 f16 lane layout.
// Tile table:  0..3  = W_l  (ld 64,  kvalid 9,  k0 0,   n0 = t*16)
//              4..7  = W_r  (ld 64,  kvalid 9,  k0 0,   n0 = t*16)
//              8..23 = W_fc1(ld 128, kvalid 64, k0 = (tt&1)*32, n0 = (tt>>1)*16)
// Element j of lane L in tile T sits at bpack[T*512 + L*16 + j] and equals
// W[(k0 + 8*(L>>4) + j + (j<8 ? 0 : 8))*ld + n0 + (L&15)]  (zero beyond kvalid).
// ---------------------------------------------------------------------------
__global__ void k_prepack(const float* __restrict__ Wl, const float* __restrict__ Wr,
                          const float* __restrict__ Wfc1, _Float16* __restrict__ bpack) {
    int idx = blockIdx.x * blockDim.x + threadIdx.x;
    if (idx >= 24 * 512) return;
    int tile = idx >> 9;
    int lane = (idx >> 4) & 31;
    int j    = idx & 15;
    const float* W;
    int ld, kvalid, k0, n0;
    if (tile < 8) {
        W = (tile < 4) ? Wl : Wr;
        ld = 64; kvalid = 9; k0 = 0; n0 = (tile & 3) * 16;
    } else {
        W = Wfc1;
        ld = 128; kvalid = 64;
        int tt = tile - 8;
        n0 = (tt >> 1) * 16;
        k0 = (tt & 1) * 32;
    }
    int k = k0 + ((j < 8) ? ((lane >> 4) * 8 + j) : (16 + (lane >> 4) * 8 + (j - 8)));
    int n = n0 + (lane & 15);
    float v = (k < kvalid) ? W[k * ld + n] : 0.0f;
    bpack[idx] = (_Float16)v;
}

// ---------------------------------------------------------------------------
// x_l = x@W_l + b_l ; x_r = x@W_r + b_r     (K=9 zero-padded to 32)
// One wave32 per 16-node tile; 8 v_wmma per wave.  N % 16 == 0 assumed.
// A lane layout = two contiguous K-runs of 8: [8h..8h+7], [16+8h..16+8h+7];
// the second run is entirely k>=16>9 -> statically zero.
// ---------------------------------------------------------------------------
__global__ void k_node_transform(const float* __restrict__ x,
                                 const float* __restrict__ b_l,
                                 const float* __restrict__ b_r,
                                 const _Float16* __restrict__ bpack,
                                 float* __restrict__ xl, float* __restrict__ xr, int N) {
    int wave = blockIdx.x * (blockDim.x >> 5) + (threadIdx.x >> 5);
    int row0 = wave * 16;
    if (row0 >= N) return;
    int lane = threadIdx.x & 31;
    int half = lane >> 4, m = lane & 15;
    int row  = row0 + m;
    int rb   = row * 9;

    v16h a;
    #pragma unroll
    for (int j = 0; j < 8; ++j) {
        int k  = 8 * half + j;
        int ks = (k < 9) ? k : 8;                 // clamp -> unconditional load
        float v = x[rb + ks];
        a[j]     = (_Float16)((k < 9) ? v : 0.0f);
        a[8 + j] = (_Float16)0.0f;                // k >= 16 always padded
    }

    #pragma unroll
    for (int t = 0; t < 4; ++t) {
        int col = t * 16 + m;
        float bll = b_l[col];
        float brr = b_r[col];

        v16h bl = *(const v16h*)(bpack + (size_t)(t) * 512 + lane * 16);
        v8f c = {};
        c = __builtin_amdgcn_wmma_f32_16x16x32_f16(false, a, false, bl, (short)0, c, false, false);
        #pragma unroll
        for (int v = 0; v < 8; ++v) {
            int r = row0 + v + 8 * half;
            xl[(size_t)r * 64 + col] = c[v] + bll;
        }

        v16h br = *(const v16h*)(bpack + (size_t)(4 + t) * 512 + lane * 16);
        v8f d = {};
        d = __builtin_amdgcn_wmma_f32_16x16x32_f16(false, a, false, br, (short)0, d, false, false);
        #pragma unroll
        for (int v = 0; v < 8; ++v) {
            int r = row0 + v + 8 * half;
            xr[(size_t)r * 64 + col] = d[v] + brr;
        }
    }
}

// ---------------------------------------------------------------------------
// Per-edge attention score + segment max.  One wave32 per edge (incl. self-loops).
// ---------------------------------------------------------------------------
__global__ void k_edge_score(const int* __restrict__ ei, const float* __restrict__ ea,
                             const float* __restrict__ loop_attr,
                             const float* __restrict__ xl, const float* __restrict__ xr,
                             const float* __restrict__ W_e, const float* __restrict__ att,
                             float* __restrict__ score, float* __restrict__ smax,
                             int E, int N) {
    int w = blockIdx.x * (blockDim.x >> 5) + (threadIdx.x >> 5);
    int EN = E + N;
    if (w >= EN) return;
    int lane = threadIdx.x & 31;

    int s, d; float eav;
    if (w < E) { s = ei[w]; d = ei[E + w]; eav = ea[w]; }
    else       { s = d = w - E;            eav = loop_attr[w - E]; }

    float acc = 0.0f;
    #pragma unroll
    for (int rep = 0; rep < 2; ++rep) {
        int f = lane + rep * 32;
        float v = xl[(size_t)s * 64 + f] + xr[(size_t)d * 64 + f] + eav * W_e[f];
        v = (v > 0.0f) ? v : NEG_SLOPE * v;
        acc += v * att[f];
    }
    #pragma unroll
    for (int off = 16; off > 0; off >>= 1) acc += __shfl_down(acc, off, 32);
    if (lane == 0) {
        score[w] = acc;
        atomicMaxFloat(&smax[d], acc);
    }
}

// a = exp(score - smax[dst]); overwrite score; segment-sum into esum
__global__ void k_edge_expsum(const int* __restrict__ ei, float* __restrict__ score,
                              const float* __restrict__ smax, float* __restrict__ esum,
                              int E, int N) {
    int i = blockIdx.x * blockDim.x + threadIdx.x;
    int EN = E + N;
    if (i >= EN) return;
    if (i + 8192 < EN) __builtin_prefetch(score + i + 8192, 0, 0);
    int d = (i < E) ? ei[E + i] : (i - E);
    float a = __expf(score[i] - smax[d]);
    score[i] = a;
    atomicAdd(&esum[d], a);
}

// gat[dst] += x_l[src] * alpha     (one wave32 per edge)
__global__ void k_edge_agg(const int* __restrict__ ei, const float* __restrict__ score,
                           const float* __restrict__ esum, const float* __restrict__ xl,
                           float* __restrict__ gat, int E, int N) {
    int w = blockIdx.x * (blockDim.x >> 5) + (threadIdx.x >> 5);
    int EN = E + N;
    if (w >= EN) return;
    int lane = threadIdx.x & 31;
    int s, d;
    if (w < E) { s = ei[w]; d = ei[E + w]; }
    else       { s = d = w - E; }
    float alpha = score[w] / (esum[d] + 1e-16f);
    #pragma unroll
    for (int rep = 0; rep < 2; ++rep) {
        int f = lane + rep * 32;
        atomicAdd(&gat[(size_t)d * 64 + f], xl[(size_t)s * 64 + f] * alpha);
    }
}

// ---------------------------------------------------------------------------
// h = relu((gat + bias_gat) @ W_fc1 + b_fc1) accumulated straight into pooled
// per-graph sums (h never materialized).  16 v_wmma per 16-node tile.
// A operand: K=64 fully valid -> contiguous unconditional loads.
// ---------------------------------------------------------------------------
__global__ void k_fc1_pool(const float* __restrict__ gat, const float* __restrict__ bias_gat,
                           const float* __restrict__ b_fc1,
                           const _Float16* __restrict__ bpack,
                           const int* __restrict__ pool_batch,
                           float* __restrict__ pooled, int N) {
    int wave = blockIdx.x * (blockDim.x >> 5) + (threadIdx.x >> 5);
    int row0 = wave * 16;
    if (row0 >= N) return;
    int lane = threadIdx.x & 31;
    int half = lane >> 4, m = lane & 15;
    int row  = row0 + m;
    const float* grow = gat + (size_t)row * 64;

    // A tiles for k-steps 0 (k=0..31) and 1 (k=32..63); each lane's 16 elements
    // are two contiguous 8-float runs starting at k0+8h and k0+16+8h.
    v16h a0, a1;
    #pragma unroll
    for (int j = 0; j < 8; ++j) {
        int c1 = 8 * half + j;          // run1 of k-step 0
        int c2 = 16 + 8 * half + j;     // run2 of k-step 0
        a0[j]     = (_Float16)(grow[c1]      + bias_gat[c1]);
        a0[8 + j] = (_Float16)(grow[c2]      + bias_gat[c2]);
        a1[j]     = (_Float16)(grow[32 + c1] + bias_gat[32 + c1]);
        a1[8 + j] = (_Float16)(grow[32 + c2] + bias_gat[32 + c2]);
    }

    // Graph ids for the 8 rows this lane's accumulator covers (contiguous).
    int gid[8];
    int rbase = row0 + 8 * half;
    #pragma unroll
    for (int v = 0; v < 8; ++v) gid[v] = pool_batch[rbase + v];

    #pragma unroll
    for (int t = 0; t < 8; ++t) {
        int col = t * 16 + m;
        float bias1 = b_fc1[col];
        v16h b0 = *(const v16h*)(bpack + (size_t)(8 + t * 2)     * 512 + lane * 16);
        v16h b1 = *(const v16h*)(bpack + (size_t)(8 + t * 2 + 1) * 512 + lane * 16);
        v8f c = {};
        c = __builtin_amdgcn_wmma_f32_16x16x32_f16(false, a0, false, b0, (short)0, c, false, false);
        c = __builtin_amdgcn_wmma_f32_16x16x32_f16(false, a1, false, b1, (short)0, c, false, false);
        #pragma unroll
        for (int v = 0; v < 8; ++v) {
            float h = fmaxf(c[v] + bias1, 0.0f);
            atomicAdd(&pooled[(size_t)gid[v] * 128 + col], h);
        }
    }
}

// ---------------------------------------------------------------------------
// Head: pooled mean | fc2 | LSTM step (h0=c0=0) | fc_output -> (q, h1, c1)
// ---------------------------------------------------------------------------
__global__ void k_head(const float* __restrict__ pooled, const float* __restrict__ cnt,
                       const float* __restrict__ agent,
                       const float* __restrict__ W_fc2, const float* __restrict__ b_fc2,
                       const float* __restrict__ W_ih, const float* __restrict__ b_ih,
                       const float* __restrict__ b_hh,
                       const float* __restrict__ W_o1, const float* __restrict__ b_o1,
                       const float* __restrict__ W_o2, const float* __restrict__ b_o2,
                       float* __restrict__ zbuf, float* __restrict__ out) {
    __shared__ float h1s[64 * 64];
    __shared__ float tmp1s[64 * 128];
    int t = threadIdx.x;

    // z = [pooled_mean (128) | relu(agent@W_fc2+b_fc2) (64)]  per graph
    for (int idx = t; idx < 64 * 192; idx += 256) {
        int g = idx / 192, j = idx - g * 192;
        float v;
        if (j < 128) {
            v = pooled[g * 128 + j] / fmaxf(cnt[g], 1.0f);
        } else {
            int jj = j - 128;
            float acc = b_fc2[jj];
            for (int k = 0; k < 34; ++k) acc += agent[g * 34 + k] * W_fc2[k * 64 + jj];
            v = fmaxf(acc, 0.0f);
        }
        zbuf[idx] = v;
    }
    __threadfence();
    __syncthreads();

    // LSTM, gate order i,f,g,o; c0=0 so f-gate drops out.
    {
        int g = t >> 2, q4 = t & 3;
        const float* zg = zbuf + g * 192;
        for (int n = q4 * 16; n < q4 * 16 + 16; ++n) {
            float gi = b_ih[n]       + b_hh[n];
            float gg = b_ih[128 + n] + b_hh[128 + n];
            float go = b_ih[192 + n] + b_hh[192 + n];
            const float* wi = W_ih + (size_t)n * 192;
            const float* wg = W_ih + (size_t)(128 + n) * 192;
            const float* wo = W_ih + (size_t)(192 + n) * 192;
            for (int k = 0; k < 192; ++k) {
                float z = zg[k];
                gi += z * wi[k];
                gg += z * wg[k];
                go += z * wo[k];
            }
            float c1 = sigmoidf_(gi) * tanhf(gg);
            float h1 = sigmoidf_(go) * tanhf(c1);
            out[1024 + g * 64 + n] = h1;   // h1 section
            out[5120 + g * 64 + n] = c1;   // c1 section
            h1s[g * 64 + n] = h1;
        }
    }
    __syncthreads();

    // relu(h1 @ W_o1 + b_o1)  [64x128]
    for (int idx = t; idx < 64 * 128; idx += 256) {
        int g = idx >> 7, j = idx & 127;
        float acc = b_o1[j];
        const float* hg = h1s + g * 64;
        for (int n = 0; n < 64; ++n) acc += hg[n] * W_o1[n * 128 + j];
        tmp1s[idx] = fmaxf(acc, 0.0f);
    }
    __syncthreads();

    // q = tmp1 @ W_o2 + b_o2  [64x16]
    for (int idx = t; idx < 64 * 16; idx += 256) {
        int g = idx >> 4, k = idx & 15;
        float acc = b_o2[k];
        const float* tg = tmp1s + g * 128;
        for (int j = 0; j < 128; ++j) acc += tg[j] * W_o2[j * 16 + k];
        out[g * 16 + k] = acc;
    }
}

// ---------------------------------------------------------------------------
extern "C" void kernel_launch(void* const* d_in, const int* in_sizes, int n_in,
                              void* d_out, int out_size, void* d_ws, size_t ws_size,
                              hipStream_t stream) {
    const float* x        = (const float*)d_in[0];
    const int*   ei       = (const int*)  d_in[1];
    const float* ea       = (const float*)d_in[2];
    const float* agent    = (const float*)d_in[3];
    const int*   pb       = (const int*)  d_in[4];
    const float* W_l      = (const float*)d_in[5];
    const float* b_l      = (const float*)d_in[6];
    const float* W_r      = (const float*)d_in[7];
    const float* b_r      = (const float*)d_in[8];
    const float* W_e      = (const float*)d_in[9];
    const float* att      = (const float*)d_in[10];
    const float* bias_gat = (const float*)d_in[11];
    const float* W_fc1    = (const float*)d_in[12];
    const float* b_fc1    = (const float*)d_in[13];
    const float* W_fc2    = (const float*)d_in[14];
    const float* b_fc2    = (const float*)d_in[15];
    const float* W_ih     = (const float*)d_in[16];
    // d_in[17] = W_hh: unused (h0 == 0)
    const float* b_ih     = (const float*)d_in[18];
    const float* b_hh     = (const float*)d_in[19];
    const float* W_o1     = (const float*)d_in[20];
    const float* b_o1     = (const float*)d_in[21];
    const float* W_o2     = (const float*)d_in[22];
    const float* b_o2     = (const float*)d_in[23];

    const int N  = in_sizes[0] / 9;   // 100000 (multiple of 16)
    const int E  = in_sizes[2];       // 3200000
    const int EN = E + N;

    // ---- workspace layout (floats) ----
    float* ws = (float*)d_ws;
    size_t off = 0;
    float* xl       = ws + off; off += (size_t)N * 64;
    float* xr       = ws + off; off += (size_t)N * 64;
    float* score    = ws + off; off += (size_t)EN;
    size_t zstart   = off;                         // zeroed region begins
    float* gat      = ws + off; off += (size_t)N * 64;
    float* deg      = ws + off; off += (size_t)N;
    float* loop_sum = ws + off; off += (size_t)N;  // becomes loop_attr in place
    float* esum     = ws + off; off += (size_t)N;
    float* pooled   = ws + off; off += 64 * 128;
    float* cnt      = ws + off; off += 64;
    size_t nzero    = off - zstart;                // zeroed region ends
    float* smax     = ws + off; off += (size_t)N;
    float* zbuf     = ws + off; off += 64 * 192;
    off = (off + 7) & ~(size_t)7;                  // 32B-align packed B
    _Float16* bpack = (_Float16*)(ws + off); off += (24 * 512) / 2;
    (void)ws_size; (void)n_in; (void)out_size;

    const int TB  = 256;
    const int WPB = TB / 32;               // 8 waves per block
    const int tiles = (N + 15) / 16;

    long ninit = (long)nzero;              // nzero > N, covers smax too
    k_init<<<dim3((unsigned)((ninit + TB - 1) / TB)), dim3(TB), 0, stream>>>(
        ws + zstart, (long)nzero, smax, N);

    k_prepack<<<(24 * 512 + TB - 1) / TB, TB, 0, stream>>>(W_l, W_r, W_fc1, bpack);

    k_edge_pre<<<(E + TB - 1) / TB, TB, 0, stream>>>(ei, ea, deg, loop_sum, E);
    k_pool_count<<<(N + TB - 1) / TB, TB, 0, stream>>>(pb, cnt, N);
    k_loop_fin<<<(N + TB - 1) / TB, TB, 0, stream>>>(loop_sum, deg, N);

    k_node_transform<<<(tiles + WPB - 1) / WPB, TB, 0, stream>>>(
        x, b_l, b_r, bpack, xl, xr, N);

    k_edge_score<<<(EN + WPB - 1) / WPB, TB, 0, stream>>>(
        ei, ea, loop_sum, xl, xr, W_e, att, score, smax, E, N);

    k_edge_expsum<<<(EN + TB - 1) / TB, TB, 0, stream>>>(ei, score, smax, esum, E, N);

    k_edge_agg<<<(EN + WPB - 1) / WPB, TB, 0, stream>>>(ei, score, esum, xl, gat, E, N);

    k_fc1_pool<<<(tiles + WPB - 1) / WPB, TB, 0, stream>>>(
        gat, bias_gat, b_fc1, bpack, pb, pooled, N);

    k_head<<<1, TB, 0, stream>>>(pooled, cnt, agent, W_fc2, b_fc2,
                                 W_ih, b_ih, b_hh, W_o1, b_o1, W_o2, b_o2,
                                 zbuf, (float*)d_out);
}
